// BnDCN_Context_25288767439681
// MI455X (gfx1250) — matrix operand order repaired and checked by reading, not compiled, not run on hardware
//
#include <hip/hip_runtime.h>
#include <hip/hip_bf16.h>
#include <math.h>

typedef _Float16 half_t;
typedef __attribute__((ext_vector_type(16))) _Float16 v16h;
typedef __attribute__((ext_vector_type(8)))  float    v8f;

#define HP      66          // padded pooled height/width
#define CCH     256
#define NB      4
#define NPOS    4096        // 64*64 pooled positions
#define PADAREA (HP*HP)     // 4356

// ---------------------------------------------------------------------------
// WMMA 16x16x32 f16 per-lane K mapping (CDNA5 ISA §7.12.2):
//  A (16x32, M x K): lane l holds row m = l&15.  half-element j maps to
//      kA(l,j) = (j&8)*2 + ((l>>4)<<3) + (j&7)
//  B (32x16, K x N): lane l holds col n = l&15; same K mapping as A:
//      kB(l,j) = (j&8)*2 + ((l>>4)<<3) + (j&7)
//  C/D (16x16 f32): lane l holds col n = l&15; element r is row m = r + 8*(l>>4)
// A tiles are pre-permuted/pre-converted to f16 in global memory once, then
// bulk-copied (b128 global load -> b128 LDS store). B tiles are built on the
// fly (bilinear gathers) directly in per-lane order.
// ---------------------------------------------------------------------------
__device__ __forceinline__ int kmapA(int l, int j) {
  return ((j & 8) << 1) + ((l >> 4) << 3) + (j & 7);
}
__device__ __forceinline__ int kmapB(int l, int j) {
  return ((j & 8) << 1) + ((l >> 4) << 3) + (j & 7);
}

// ===========================================================================
// 0a) one-shot: permute dcn_w (256,256,3,3) f32 -> f16 A-tile image
//     layout: [ot(2)][kc(72)][s(8)][lane(32)][j(16)]  (k = tap*256 + c)
// ===========================================================================
__global__ __launch_bounds__(256) void k_wperm_dcn(const float* __restrict__ w,
                                                   half_t* __restrict__ aw) {
  int t = blockIdx.x * 256 + threadIdx.x;            // 589,824 elements
  int j = t & 15, l = (t >> 4) & 31, s = (t >> 9) & 7;
  int blk = t >> 12;                                  // 0..143
  int kc = blk % 72, ot = blk / 72;
  int o = ot * 128 + s * 16 + (l & 15);
  int c = (kc & 7) * 32 + kmapA(l, j);
  int n = kc >> 3;
  aw[t] = (half_t)w[(size_t)o * 2304 + (size_t)c * 9 + n];
}

// ===========================================================================
// 0b) one-shot: permute f_w (256,512) f32 -> f16 A-tile image
//     layout: [ot(2)][kc(16)][s(8)][lane(32)][j(16)]
// ===========================================================================
__global__ __launch_bounds__(256) void k_wperm_fin(const float* __restrict__ w,
                                                   half_t* __restrict__ aw) {
  int t = blockIdx.x * 256 + threadIdx.x;            // 131,072 elements
  int j = t & 15, l = (t >> 4) & 31, s = (t >> 9) & 7;
  int blk = t >> 12;                                  // 0..31
  int kc = blk % 16, ot = blk / 16;
  int o = ot * 128 + s * 16 + (l & 15);
  int k = kc * 32 + kmapA(l, j);
  aw[t] = (half_t)w[(size_t)o * 512 + k];
}

// ===========================================================================
// 1) 2x2 maxpool (4,256,128,128) -> zero-padded pooled image (4,256,66,66)
// ===========================================================================
__global__ __launch_bounds__(256) void k_pool(const float* __restrict__ x,
                                              float* __restrict__ xpad) {
  size_t t = (size_t)blockIdx.x * 256 + threadIdx.x;   // over 4*256*66*66
  int ij = (int)(t % PADAREA);
  size_t bc = t / PADAREA;
  int i = ij / HP, j = ij % HP;
  float v = 0.f;
  if (i >= 1 && i <= 64 && j >= 1 && j <= 64) {
    const float* p = x + bc * (128 * 128) + (size_t)((i - 1) * 2) * 128 + (j - 1) * 2;
    v = fmaxf(fmaxf(p[0], p[1]), fmaxf(p[128], p[129]));
  }
  xpad[bc * PADAREA + ij] = v;
}

// ===========================================================================
// 2) fused offset/mod 3x3 convs + sigmoid + bilinear coefficient precompute.
//    Block = (row, batch). 256 threads: col = tid&63, channel-group = tid>>6.
//    Emits per (b,p,tap): int4 gather indices + float4 mod-premultiplied wts.
// ===========================================================================
__global__ __launch_bounds__(256) void k_prep(const float* __restrict__ xpad,
                                              const float* __restrict__ p_w,
                                              const float* __restrict__ p_b,
                                              const float* __restrict__ m_w,
                                              const float* __restrict__ m_b,
                                              int4* __restrict__ sidx,
                                              float4* __restrict__ swt) {
  __shared__ float part[4][64][27];
  const int tid = threadIdx.x;
  const int row = blockIdx.x;
  const int b   = blockIdx.y;
  const int col = tid & 63;
  const int cg  = tid >> 6;

  const float* Xb = xpad + (size_t)b * CCH * PADAREA;
  float acc[27];
#pragma unroll
  for (int i = 0; i < 27; ++i) acc[i] = 0.f;

  for (int c = cg * 64; c < cg * 64 + 64; ++c) {
    const float* Xc = Xb + (size_t)c * PADAREA + row * HP + col;
    float xw[9] = {Xc[0], Xc[1], Xc[2], Xc[66], Xc[67], Xc[68], Xc[132], Xc[133], Xc[134]};
#pragma unroll
    for (int oc = 0; oc < 18; ++oc) {
      const float* w = p_w + ((size_t)oc * CCH + c) * 9;
#pragma unroll
      for (int tt = 0; tt < 9; ++tt) acc[oc] += w[tt] * xw[tt];
    }
#pragma unroll
    for (int oc = 0; oc < 9; ++oc) {
      const float* w = m_w + ((size_t)oc * CCH + c) * 9;
#pragma unroll
      for (int tt = 0; tt < 9; ++tt) acc[18 + oc] += w[tt] * xw[tt];
    }
  }
#pragma unroll
  for (int i = 0; i < 27; ++i) part[cg][col][i] = acc[i];
  __syncthreads();

  if (tid < 64) {
    const int c2 = tid;
    const float pnx[9] = {-1, -1, -1, 0, 0, 0, 1, 1, 1};
    const float pny[9] = {-1, 0, 1, -1, 0, 1, -1, 0, 1};
#pragma unroll
    for (int n = 0; n < 9; ++n) {
      float sx = part[0][c2][n] + part[1][c2][n] + part[2][c2][n] + part[3][c2][n] + p_b[n];
      float sy = part[0][c2][9 + n] + part[1][c2][9 + n] + part[2][c2][9 + n] + part[3][c2][9 + n] + p_b[9 + n];
      float sm = part[0][c2][18 + n] + part[1][c2][18 + n] + part[2][c2][18 + n] + part[3][c2][18 + n] + m_b[n];
      float mm = 1.f / (1.f + __expf(-sm));
      float px = sx + pnx[n] + (float)(row + 1);      // padded coords
      float py = sy + pny[n] + (float)(c2 + 1);
      float qlx = fminf(fmaxf(floorf(px), 0.f), 65.f);
      float qly = fminf(fmaxf(floorf(py), 0.f), 65.f);
      float qrx = fminf(qlx + 1.f, 65.f);
      float qry = fminf(qly + 1.f, 65.f);
      float pxc = fminf(fmaxf(px, 0.f), 65.f);
      float pyc = fminf(fmaxf(py, 0.f), 65.f);
      float glt = (1.f + (qlx - pxc)) * (1.f + (qly - pyc));
      float grb = (1.f - (qrx - pxc)) * (1.f - (qry - pyc));
      float glb = (1.f + (qlx - pxc)) * (1.f - (qry - pyc));
      float grt = (1.f - (qrx - pxc)) * (1.f + (qly - pyc));
      int ilx = (int)qlx, ily = (int)qly, irx = (int)qrx, iry = (int)qry;
      int base = ((b * NPOS + row * 64 + c2) * 9 + n);
      sidx[base] = make_int4(ilx * HP + ily, irx * HP + iry, ilx * HP + iry, irx * HP + ily);
      swt[base]  = make_float4(glt * mm, grb * mm, glb * mm, grt * mm);
    }
  }
}

// ===========================================================================
// 3) DCN einsum via WMMA. Block = (128 out-ch) x (one 64-pixel image row).
//    8 waves: wave w owns o-subtile w and multiplies against all 4 position
//    subtiles (4 accumulators, 4 WMMAs per K chunk). A tiles arrive as bulk
//    b128 copies of the pre-permuted f16 image; B tiles are deform-sampled
//    on the fly. K = 2304 = 72 chunks of 32 (k = tap*256 + c).
// ===========================================================================
__global__ __launch_bounds__(256) void k_dcn(const float* __restrict__ xpad,
                                             const half_t* __restrict__ awA,
                                             const float* __restrict__ dcn_b,
                                             const int4* __restrict__ sidx,
                                             const float4* __restrict__ swt,
                                             float* __restrict__ ypre) {
  __shared__ alignas(32) half_t aT[8][32][16];   // 8 o-subtiles, per-lane order
  __shared__ alignas(32) half_t bT[4][32][16];   // 4 p-subtiles, per-lane order
  __shared__ int4   sIdx[64 * 9];
  __shared__ float4 sWt[64 * 9];

  const int tid  = threadIdx.x;
  const int lane = tid & 31;
  const int wv   = tid >> 5;          // 0..7 = o-subtile owned by this wave
  const int o_base = blockIdx.x * 128;
  const int row  = blockIdx.y;
  const int b    = blockIdx.z;
  const int prow = row * 64;

  for (int e = tid; e < 64 * 9; e += 256) {
    int g = (b * NPOS + prow + (e / 9)) * 9 + (e % 9);
    sIdx[e] = sidx[g];
    sWt[e]  = swt[g];
  }

  v8f acc[4];
#pragma unroll
  for (int q = 0; q < 4; ++q) acc[q] = (v8f){0, 0, 0, 0, 0, 0, 0, 0};
  const float* Xb = xpad + (size_t)b * CCH * PADAREA;
  const uint4* awA4 = (const uint4*)awA;                 // 512 uint4 per chunk
  uint4* aFlat = (uint4*)&aT[0][0][0];

  for (int kc = 0; kc < 72; ++kc) {
    const int n  = kc >> 3;
    const int c0 = (kc & 7) * 32;
    const uint4* src = awA4 + (size_t)(blockIdx.x * 72 + kc) * 512;
    __syncthreads();
    // stage A: bulk copy of pre-permuted f16 weight tile (2 x b128 per thread)
    aFlat[tid]       = src[tid];
    aFlat[tid + 256] = src[tid + 256];
    // prefetch next chunk's weight tile
    if (kc + 1 < 72) __builtin_prefetch((const void*)(src + 512 + tid * 2), 0, 1);
    // stage B: modulated bilinear samples, permuted per-lane (8 elems/thread)
#pragma unroll
    for (int i = 0; i < 8; ++i) {
      int e = tid + i * 256;
      int j = e & 15, l = (e >> 4) & 31, s = e >> 9;
      int kk = kmapB(l, j);
      int pl = s * 16 + (l & 15);
      int4 q = sIdx[pl * 9 + n];
      float4 w = sWt[pl * 9 + n];
      const float* Xc = Xb + (size_t)(c0 + kk) * PADAREA;
      float v = w.x * Xc[q.x] + w.y * Xc[q.y] + w.z * Xc[q.z] + w.w * Xc[q.w];
      bT[s][l][j] = (half_t)v;
    }
    __syncthreads();
    v16h av = *(const v16h*)(&aT[wv][lane][0]);
#pragma unroll
    for (int q = 0; q < 4; ++q) {
      v16h bv = *(const v16h*)(&bT[q][lane][0]);
      acc[q] = __builtin_amdgcn_wmma_f32_16x16x32_f16(false, av, false, bv, (short)0, acc[q], false, false);
    }
  }

#pragma unroll
  for (int r = 0; r < 8; ++r) {
    int o = o_base + wv * 16 + ((lane >> 4) << 3) + r;
    float bias = dcn_b[o];
#pragma unroll
    for (int q = 0; q < 4; ++q) {
      int p = prow + q * 16 + (lane & 15);
      ypre[((size_t)(b * CCH + o)) * NPOS + p] = acc[q][r] + bias;
    }
  }
}

// ===========================================================================
// 4) BN stats over (b,h,w) per channel -> folded scale/shift
// ===========================================================================
__global__ __launch_bounds__(256) void k_bnstats(const float* __restrict__ ypre,
                                                 const float* __restrict__ bn_g,
                                                 const float* __restrict__ bn_b,
                                                 float* __restrict__ bnA,
                                                 float* __restrict__ bnB) {
  __shared__ float s1[256], s2[256];
  const int c = blockIdx.x;
  float a = 0.f, q = 0.f;
  for (int i = threadIdx.x; i < NB * NPOS; i += 256) {
    int b = i >> 12, p = i & 4095;
    float v = ypre[((size_t)(b * CCH + c)) * NPOS + p];
    a += v; q += v * v;
  }
  s1[threadIdx.x] = a; s2[threadIdx.x] = q;
  __syncthreads();
  for (int st = 128; st > 0; st >>= 1) {
    if (threadIdx.x < st) { s1[threadIdx.x] += s1[threadIdx.x + st]; s2[threadIdx.x] += s2[threadIdx.x + st]; }
    __syncthreads();
  }
  if (threadIdx.x == 0) {
    float mu  = s1[0] * (1.f / 16384.f);
    float var = s2[0] * (1.f / 16384.f) - mu * mu;
    float aa = bn_g[c] * rsqrtf(var + 1e-5f);
    bnA[c] = aa;
    bnB[c] = bn_b[c] - aa * mu;
  }
}

// ===========================================================================
// 5) global-context path: mask conv -> softmax -> ctx -> FC -> LN -> FC
//    One block per batch; attention kept in LDS.
// ===========================================================================
__global__ __launch_bounds__(256) void k_ctx(const float* __restrict__ xpad,
                                             const float* __restrict__ cm_w,
                                             const float* __restrict__ cm_b,
                                             const float* __restrict__ c1_w,
                                             const float* __restrict__ c1_b,
                                             const float* __restrict__ ln_g,
                                             const float* __restrict__ ln_b,
                                             const float* __restrict__ c2_w,
                                             const float* __restrict__ c2_b,
                                             float* __restrict__ tvec) {
  __shared__ float smask[4096];
  __shared__ float red[256];
  __shared__ float sctx[256];
  __shared__ float st[64];
  __shared__ float sstat[2];
  const int b = blockIdx.x;
  const int tid = threadIdx.x;
  const float* Xb = xpad + (size_t)b * CCH * PADAREA;

  for (int p = tid; p < 4096; p += 256) {
    int r = p >> 6, cx = p & 63;
    int off = (r + 1) * HP + (cx + 1);
    float acc = cm_b[0];
    for (int c = 0; c < CCH; ++c) acc += cm_w[c] * Xb[(size_t)c * PADAREA + off];
    smask[p] = acc;
  }
  __syncthreads();
  float m = -1e30f;
  for (int p = tid; p < 4096; p += 256) m = fmaxf(m, smask[p]);
  red[tid] = m; __syncthreads();
  for (int s = 128; s > 0; s >>= 1) { if (tid < s) red[tid] = fmaxf(red[tid], red[tid + s]); __syncthreads(); }
  m = red[0]; __syncthreads();
  float sum = 0.f;
  for (int p = tid; p < 4096; p += 256) { float e = __expf(smask[p] - m); smask[p] = e; sum += e; }
  red[tid] = sum; __syncthreads();
  for (int s = 128; s > 0; s >>= 1) { if (tid < s) red[tid] += red[tid + s]; __syncthreads(); }
  float inv = 1.f / red[0];
  __syncthreads();
  for (int p = tid; p < 4096; p += 256) smask[p] *= inv;
  __syncthreads();
  { // ctx[c] = sum_p x[c,p] * attn[p]
    const float* Xc = Xb + (size_t)tid * PADAREA;
    float acc = 0.f;
    for (int p = 0; p < 4096; ++p) {
      int r = p >> 6, cx = p & 63;
      acc += smask[p] * Xc[(r + 1) * HP + cx + 1];
    }
    sctx[tid] = acc;
  }
  __syncthreads();
  if (tid < 64) {
    float acc = c1_b[tid];
    for (int c = 0; c < CCH; ++c) acc += c1_w[tid * CCH + c] * sctx[c];
    st[tid] = acc;
  }
  __syncthreads();
  if (tid == 0) {
    float mu = 0.f;
    for (int r = 0; r < 64; ++r) mu += st[r];
    mu *= (1.f / 64.f);
    float var = 0.f;
    for (int r = 0; r < 64; ++r) { float d = st[r] - mu; var += d * d; }
    var *= (1.f / 64.f);
    sstat[0] = mu; sstat[1] = rsqrtf(var + 1e-5f);
  }
  __syncthreads();
  if (tid < 64) {
    float v = ln_g[tid] * (st[tid] - sstat[0]) * sstat[1] + ln_b[tid];
    st[tid] = fmaxf(v, 0.f);
  }
  __syncthreads();
  {
    float acc = c2_b[tid];
    for (int r = 0; r < 64; ++r) acc += c2_w[tid * 64 + r] * st[r];
    tvec[b * CCH + tid] = acc;
  }
}

// ===========================================================================
// 6) fused BN+ReLU | (x+t) concat -> 1x1 conv (WMMA, K=512) + f_b + residual x
//    Same 128-channel x 64-position tiling as k_dcn; A from pre-permuted f_w.
// ===========================================================================
__global__ __launch_bounds__(256) void k_final(const float* __restrict__ xpad,
                                               const float* __restrict__ ypre,
                                               const float* __restrict__ bnA,
                                               const float* __restrict__ bnB,
                                               const float* __restrict__ tvec,
                                               const half_t* __restrict__ awF,
                                               const float* __restrict__ f_b,
                                               float* __restrict__ out) {
  __shared__ alignas(32) half_t aT[8][32][16];
  __shared__ alignas(32) half_t bT[4][32][16];
  const int tid  = threadIdx.x;
  const int lane = tid & 31;
  const int wv   = tid >> 5;
  const int o_base = blockIdx.x * 128;
  const int row  = blockIdx.y;
  const int b    = blockIdx.z;
  const int prow = row * 64;

  v8f acc[4];
#pragma unroll
  for (int q = 0; q < 4; ++q) acc[q] = (v8f){0, 0, 0, 0, 0, 0, 0, 0};
  const float* Xb = xpad + (size_t)b * CCH * PADAREA;
  const uint4* awF4 = (const uint4*)awF;
  uint4* aFlat = (uint4*)&aT[0][0][0];

  for (int kc = 0; kc < 16; ++kc) {
    const int k0 = kc * 32;
    const uint4* src = awF4 + (size_t)(blockIdx.x * 16 + kc) * 512;
    __syncthreads();
    aFlat[tid]       = src[tid];
    aFlat[tid + 256] = src[tid + 256];
    if (kc + 1 < 16) __builtin_prefetch((const void*)(src + 512 + tid * 2), 0, 1);
#pragma unroll
    for (int i = 0; i < 8; ++i) {
      int e = tid + i * 256;
      int j = e & 15, l = (e >> 4) & 31, s = e >> 9;
      int k = k0 + kmapB(l, j);
      int p = prow + s * 16 + (l & 15);
      float v;
      if (k < 256) {        // y branch: BN + ReLU
        v = fmaxf(bnA[k] * ypre[((size_t)(b * CCH + k)) * NPOS + p] + bnB[k], 0.f);
      } else {              // z branch: x + t broadcast
        int c = k - 256;
        int r = p >> 6, cx = p & 63;
        v = Xb[(size_t)c * PADAREA + (r + 1) * HP + cx + 1] + tvec[b * CCH + c];
      }
      bT[s][l][j] = (half_t)v;
    }
    __syncthreads();
    v16h av = *(const v16h*)(&aT[wv][lane][0]);
#pragma unroll
    for (int q = 0; q < 4; ++q) {
      v16h bv = *(const v16h*)(&bT[q][lane][0]);
      acc[q] = __builtin_amdgcn_wmma_f32_16x16x32_f16(false, av, false, bv, (short)0, acc[q], false, false);
    }
  }

#pragma unroll
  for (int r = 0; r < 8; ++r) {
    int o = o_base + wv * 16 + ((lane >> 4) << 3) + r;
    float fb = f_b[o];
#pragma unroll
    for (int q = 0; q < 4; ++q) {
      int p = prow + q * 16 + (lane & 15);
      int rr = p >> 6, cx = p & 63;
      float xv = Xb[(size_t)o * PADAREA + (rr + 1) * HP + cx + 1];
      out[((size_t)(b * CCH + o)) * NPOS + p] = acc[q][r] + fb + xv;
    }
  }
}

// ===========================================================================
extern "C" void kernel_launch(void* const* d_in, const int* in_sizes, int n_in,
                              void* d_out, int out_size, void* d_ws, size_t ws_size,
                              hipStream_t stream) {
  const float* x     = (const float*)d_in[0];
  const float* p_w   = (const float*)d_in[1];
  const float* p_b   = (const float*)d_in[2];
  const float* m_w   = (const float*)d_in[3];
  const float* m_b   = (const float*)d_in[4];
  const float* dcn_w = (const float*)d_in[5];
  const float* dcn_b = (const float*)d_in[6];
  const float* bn_g  = (const float*)d_in[7];
  const float* bn_b  = (const float*)d_in[8];
  const float* cm_w  = (const float*)d_in[9];
  const float* cm_b  = (const float*)d_in[10];
  const float* c1_w  = (const float*)d_in[11];
  const float* c1_b  = (const float*)d_in[12];
  const float* ln_g  = (const float*)d_in[13];
  const float* ln_b  = (const float*)d_in[14];
  const float* c2_w  = (const float*)d_in[15];
  const float* c2_b  = (const float*)d_in[16];
  const float* f_w   = (const float*)d_in[17];
  const float* f_b   = (const float*)d_in[18];
  float* out = (float*)d_out;

  char* ws = (char*)d_ws;
  float*  xpad = (float*)(ws);                   // 4*256*66*66 f32 = 17,842,176 B
  float*  ypre = (float*)(ws + 17842176);        // 4*256*4096 f32  = 16,777,216 B
  int4*   sidx = (int4* )(ws + 34619392);        // 4*4096*9 int4   =  2,359,296 B
  float4* swt  = (float4*)(ws + 36978688);       // 4*4096*9 float4 =  2,359,296 B
  float*  bnA  = (float*)(ws + 39337984);        // 256 f32 (padded slot)
  float*  bnB  = (float*)(ws + 39342080);        // 256 f32 (padded slot)
  float*  tvec = (float*)(ws + 39346176);        // 4*256 f32 (padded slot)
  half_t* awA  = (half_t*)(ws + 39350272);       // 589,824 f16 = 1,179,648 B
  half_t* awF  = (half_t*)(ws + 40529920);       // 131,072 f16 =   262,144 B
                                                 // total ws used ~ 40.8 MB

  k_wperm_dcn<<<2304, 256, 0, stream>>>(dcn_w, awA);
  k_wperm_fin<<<512, 256, 0, stream>>>(f_w, awF);
  k_pool<<<17424, 256, 0, stream>>>(x, xpad);
  k_prep<<<dim3(64, 4), 256, 0, stream>>>(xpad, p_w, p_b, m_w, m_b, sidx, swt);
  k_dcn<<<dim3(2, 64, 4), 256, 0, stream>>>(xpad, awA, dcn_b, sidx, swt, ypre);
  k_bnstats<<<256, 256, 0, stream>>>(ypre, bn_g, bn_b, bnA, bnB);
  k_ctx<<<4, 256, 0, stream>>>(xpad, cm_w, cm_b, c1_w, c1_b, ln_g, ln_b, c2_w, c2_b, tvec);
  k_final<<<dim3(2, 64, 4), 256, 0, stream>>>(xpad, ypre, bnA, bnB, tvec, awF, f_b, out);
}
